// Pointsformer2_65309272703076
// MI455X (gfx1250) — compile-verified, matrix operand order checked by reading, not run on hardware
//
#include <hip/hip_runtime.h>
#include <cstdint>
#include <cstddef>

// ---------------------------------------------------------------------------
// MI455X (gfx1250) Pointsformer forward.
// Core math runs on v_wmma_f32_16x16x32_f16 (wave32 WMMA, f32 accum).
// Attention (qkv proj + flash softmax + A*V + Wo proj) fully fused in LDS.
// GEMM operand staging uses CDNA5 async global->LDS copies (ASYNCcnt).
// ---------------------------------------------------------------------------

typedef __attribute__((ext_vector_type(16))) _Float16 v16h;
typedef __attribute__((ext_vector_type(8)))  float    v8f;

#define DEVI __device__ __forceinline__

DEVI int lane_id() { return threadIdx.x & 31; }

DEVI v8f v8zero() {
  v8f c;
#pragma unroll
  for (int r = 0; r < 8; ++r) c[r] = 0.f;
  return c;
}

// K-index of element i of an A/B fragment for lane l (ISA 7.12.2, 16-bit, 16x32).
DEVI int frag_k(int i, int l) {
  const int j = i >> 1;
  return ((j & 4) ? 16 : 0) + (((l >> 4) & 1) << 3) + ((j & 3) << 1) + (i & 1);
}

// A fragment (M x K row-major, f32 in LDS -> f16 regs)
DEVI v16h frag_a_f32(const float* s, int row0, int ld) {
  const int l = lane_id(); const int m = row0 + (l & 15);
  v16h f;
#pragma unroll
  for (int i = 0; i < 16; ++i) f[i] = (_Float16)s[(size_t)m * ld + frag_k(i, l)];
  return f;
}
// B fragment (K x N row-major, f32 in LDS)
DEVI v16h frag_b_f32(const float* s, int col0, int ld) {
  const int l = lane_id(); const int n = col0 + (l & 15);
  v16h f;
#pragma unroll
  for (int i = 0; i < 16; ++i) f[i] = (_Float16)s[(size_t)frag_k(i, l) * ld + n];
  return f;
}
DEVI v16h frag_a_f16(const _Float16* s, int row0, int ld) {
  const int l = lane_id(); const int m = row0 + (l & 15);
  v16h f;
#pragma unroll
  for (int i = 0; i < 16; ++i) f[i] = s[(size_t)m * ld + frag_k(i, l)];
  return f;
}
// B fragment, storage K x N row-major (f16)
DEVI v16h frag_b_f16(const _Float16* s, int col0, int ld) {
  const int l = lane_id(); const int n = col0 + (l & 15);
  v16h f;
#pragma unroll
  for (int i = 0; i < 16; ++i) f[i] = s[(size_t)frag_k(i, l) * ld + n];
  return f;
}
// B fragment, storage N x K row-major (i.e. B given transposed; B[k][n] = s[n*ld + k])
DEVI v16h frag_b_f16_T(const _Float16* s, int col0, int ld) {
  const int l = lane_id(); const int n = col0 + (l & 15);
  v16h f;
#pragma unroll
  for (int i = 0; i < 16; ++i) f[i] = s[(size_t)n * ld + frag_k(i, l)];
  return f;
}

DEVI v8f wmma_f16(v16h a, v16h b, v8f c) {
  return __builtin_amdgcn_wmma_f32_16x16x32_f16(false, a, false, b, (short)0, c,
                                                false, false);
}

// CDNA5 async global -> LDS copy (tracked by ASYNCcnt).
DEVI uint32_t lds_off_of(const void* p) { return (uint32_t)(uintptr_t)p; }
DEVI void async_b128(uint32_t lds_off, const float* g) {
  asm volatile("global_load_async_to_lds_b128 %0, %1, off"
               :: "v"(lds_off), "v"(g) : "memory");
}
DEVI void wait_async0() { asm volatile("s_wait_asynccnt 0x0" ::: "memory"); }

// ---------------------------------------------------------------------------
// Generic batched GEMM: Y[b] = W[M,K] @ X[b][K,N]  (+bias, +relu)
// 4 waves / block, 64x32 output tile, K staged in 32-chunks via async copies.
// ---------------------------------------------------------------------------
#define GF_BIAS 1
#define GF_RELU 2

__global__ __launch_bounds__(128) void k_wmma_gemm(
    float* __restrict__ Y, const float* __restrict__ X, const float* __restrict__ W,
    const float* __restrict__ bias, int M, int K, int N, int flags)
{
  __shared__ __align__(16) float Ws[64 * 32];
  __shared__ __align__(16) float Xs[32 * 32];
  const int b = blockIdx.x, m0 = blockIdx.y * 64, n0 = blockIdx.z * 32;
  const float* Xb = X + (size_t)b * K * N;
  float* Yb = Y + (size_t)b * M * N;
  const int t = threadIdx.x, w = t >> 5;
  v8f c0 = v8zero(), c1 = v8zero();

  for (int k0 = 0; k0 < K; k0 += 32) {
    const int kv = (K - k0 < 32) ? (K - k0) : 32;
    if (kv == 32) {
      if (k0 + 32 < K) __builtin_prefetch(Xb + (size_t)(k0 + 32) * N + n0, 0, 1);
      // stage W tile [64][32]
      for (int j = t; j < 512; j += 128) {
        const int row = j >> 3, seg = j & 7;
        float* dst = &Ws[row * 32 + seg * 4];
        const int gm = m0 + row;
        if (gm < M) async_b128(lds_off_of(dst), W + (size_t)gm * K + k0 + seg * 4);
        else { dst[0] = 0.f; dst[1] = 0.f; dst[2] = 0.f; dst[3] = 0.f; }
      }
      // stage X tile [32][32]
      for (int j = t; j < 256; j += 128) {
        const int row = j >> 3, seg = j & 7;
        async_b128(lds_off_of(&Xs[row * 32 + seg * 4]),
                   Xb + (size_t)(k0 + row) * N + n0 + seg * 4);
      }
      wait_async0();
    } else {
      for (int j = t; j < 64 * 32; j += 128) {
        const int row = j >> 5, kk = j & 31;
        const int gm = m0 + row;
        Ws[j] = (gm < M && kk < kv) ? W[(size_t)gm * K + k0 + kk] : 0.f;
      }
      for (int j = t; j < 32 * 32; j += 128) {
        const int row = j >> 5, nn = j & 31;
        Xs[j] = (row < kv) ? Xb[(size_t)(k0 + row) * N + n0 + nn] : 0.f;
      }
    }
    __syncthreads();
    v16h a  = frag_a_f32(Ws, w * 16, 32);
    v16h b0 = frag_b_f32(Xs, 0, 32);
    v16h b1 = frag_b_f32(Xs, 16, 32);
    c0 = wmma_f16(a, b0, c0);
    c1 = wmma_f16(a, b1, c1);
    __syncthreads();
  }
  const int l = lane_id();
  const int mb = m0 + w * 16 + ((l >> 4) << 3);
  const int n = n0 + (l & 15);
#pragma unroll
  for (int r = 0; r < 8; ++r) {
    const int m = mb + r;
    if (m < M) {
      float y0 = c0[r], y1 = c1[r];
      if (flags & GF_BIAS) { const float bb = bias[m]; y0 += bb; y1 += bb; }
      if (flags & GF_RELU) { y0 = fmaxf(y0, 0.f); y1 = fmaxf(y1, 0.f); }
      Yb[(size_t)m * N + n] = y0;
      Yb[(size_t)m * N + n + 16] = y1;
    }
  }
}

// ---------------------------------------------------------------------------
// Fully-fused attention: per (seq, head) workgroup.
//   xh = x + pe (head slice, K padded to 32)
//   q/k/v = Wq/k/v @ xh  (WMMA, q pre-scaled by HD^-0.5 = 1/8)
//   flash softmax over 32-row tiles, O = P @ V (WMMA)
//   Y = Wo @ O^T + bo  (WMMA) -> raw pre-BN output
// Everything stays in LDS (<= ~283KB at L=512, within 320KB WGP LDS).
// ---------------------------------------------------------------------------
__global__ __launch_bounds__(128) void k_attn_fused(
    float* __restrict__ Y, const float* __restrict__ Xf, const float* __restrict__ Pe,
    const float* __restrict__ Wq, const float* __restrict__ Wk, const float* __restrict__ Wv,
    const float* __restrict__ Wo, const float* __restrict__ Bo, int C, int L)
{
  extern __shared__ __align__(16) char smem_raw[];
  const int seq = blockIdx.x, head = blockIdx.y;
  const int ipg = C >> 4;                       // in-channels per head (<= 32)
  const int t = threadIdx.x, w = t >> 5, l = lane_id();

  float* xs      = (float*)smem_raw;            // [32][L]  (x+pe, K-padded)
  float* sS      = xs + 32 * L;                 // [32][32] score tile
  float* rowm    = sS + 32 * 32;                // [32] running max
  float* rowsum  = rowm + 32;                   // [32] running sum
  float* rscale  = rowsum + 32;                 // [32] rescale factor
  _Float16* qs   = (_Float16*)(rscale + 32);    // [L][64]
  _Float16* ks   = qs + (size_t)L * 64;         // [L][64]
  _Float16* vs   = ks + (size_t)L * 64;         // [L][64]
  _Float16* wqs  = vs + (size_t)L * 64;         // [64][32]
  _Float16* wks  = wqs + 64 * 32;
  _Float16* wvs  = wks + 64 * 32;
  _Float16* wos  = wvs + 64 * 32;               // [32][64]
  _Float16* sP   = wos + 32 * 64;               // [32][32] probs (f16)
  _Float16* ot   = sP + 32 * 32;                // [32][64] O tile (f16)

  // ---- stage weights (zero-padded) and x+pe slice ----
  for (int j = t; j < 64 * 32; j += 128) {
    const int m = j >> 5, k = j & 31;
    float vq = 0.f, vk = 0.f, vv = 0.f;
    if (k < ipg) {
      const size_t o = ((size_t)head * 64 + m) * ipg + k;
      vq = Wq[o]; vk = Wk[o]; vv = Wv[o];
    }
    wqs[j] = (_Float16)(vq * 0.125f);   // fold HD^-0.5 into Wq
    wks[j] = (_Float16)vk;
    wvs[j] = (_Float16)vv;
  }
  for (int j = t; j < 32 * 64; j += 128) {
    const int m = j >> 6, k = j & 63;
    wos[j] = (_Float16)((m < ipg) ? Wo[((size_t)head * ipg + m) * 64 + k] : 0.f);
  }
  for (int j = t; j < 32 * L; j += 128) {
    const int i = j / L, ll = j % L;
    float v = 0.f;
    if (i < ipg) {
      const size_t o = ((size_t)seq * C + head * ipg + i) * L + ll;
      v = Xf[o] + Pe[o];
    }
    xs[j] = v;
  }
  __syncthreads();

  // ---- q/k/v projections: [64][L] = W[64][32] @ xs[32][L]; store transposed [L][64] ----
  const int ntiles = L >> 4;
  for (int tile = w; tile < 4 * ntiles; tile += 4) {
    const int mt = tile & 3, nt = tile >> 2;
    v16h bx = frag_b_f32(xs, nt * 16, L);
    v8f cq = v8zero(), ck = v8zero(), cv = v8zero();
    cq = wmma_f16(frag_a_f16(wqs, mt * 16, 32), bx, cq);
    ck = wmma_f16(frag_a_f16(wks, mt * 16, 32), bx, ck);
    cv = wmma_f16(frag_a_f16(wvs, mt * 16, 32), bx, cv);
    const int n = nt * 16 + (l & 15);
    const int mb = mt * 16 + ((l >> 4) << 3);
#pragma unroll
    for (int r = 0; r < 8; ++r) {
      qs[(size_t)n * 64 + mb + r] = (_Float16)cq[r];
      ks[(size_t)n * 64 + mb + r] = (_Float16)ck[r];
      vs[(size_t)n * 64 + mb + r] = (_Float16)cv[r];
    }
  }
  __syncthreads();

  // ---- flash attention over 32-row tiles ----
  for (int r0 = 0; r0 < L; r0 += 32) {
    if (t < 32) { rowm[t] = -3.0e30f; rowsum[t] = 0.f; }
    v8f o0 = v8zero(), o1 = v8zero();
    const int wr = (w >> 1) * 16, wc = (w & 1) * 32;
    __syncthreads();
    for (int cb = 0; cb < L; cb += 32) {
      // S tile (wave w -> rows sr, cols sc), K = 64 in 2 chunks
      const int sr = (w >> 1) * 16, sc = (w & 1) * 16;
      v8f s = v8zero();
      s = wmma_f16(frag_a_f16(qs + (size_t)r0 * 64, sr, 64),
                   frag_b_f16_T(ks + (size_t)cb * 64, sc, 64), s);
      s = wmma_f16(frag_a_f16(qs + (size_t)r0 * 64 + 32, sr, 64),
                   frag_b_f16_T(ks + (size_t)cb * 64 + 32, sc, 64), s);
      {
        const int n = sc + (l & 15), mb = sr + ((l >> 4) << 3);
#pragma unroll
        for (int r = 0; r < 8; ++r) sS[(mb + r) * 32 + n] = s[r];
      }
      __syncthreads();
      // online softmax (one wave: 32 rows)
      if (t < 32) {
        const float mold = rowm[t];
        float mx = mold;
        for (int j2 = 0; j2 < 32; ++j2) mx = fmaxf(mx, sS[t * 32 + j2]);
        const float sc2 = __expf(mold - mx);
        float sum = rowsum[t] * sc2;
        for (int j2 = 0; j2 < 32; ++j2) {
          const float p = __expf(sS[t * 32 + j2] - mx);
          sP[t * 32 + j2] = (_Float16)p;
          sum += p;
        }
        rowm[t] = mx; rowsum[t] = sum; rscale[t] = sc2;
      }
      __syncthreads();
      // rescale accumulators, accumulate O += P @ V
      {
        const int mb = wr + ((l >> 4) << 3);
#pragma unroll
        for (int r = 0; r < 8; ++r) { const float f = rscale[mb + r]; o0[r] *= f; o1[r] *= f; }
      }
      v16h ap = frag_a_f16(sP, wr, 32);
      o0 = wmma_f16(ap, frag_b_f16(vs + (size_t)cb * 64, wc, 64), o0);
      o1 = wmma_f16(ap, frag_b_f16(vs + (size_t)cb * 64, wc + 16, 64), o1);
      __syncthreads();
    }
    // normalize, dump O tile (f16) for the Wo projection
    {
      const int mb = wr + ((l >> 4) << 3);
      const int n = (l & 15);
#pragma unroll
      for (int r = 0; r < 8; ++r) {
        const float inv = 1.f / rowsum[mb + r];
        ot[(mb + r) * 64 + wc + n]      = (_Float16)(o0[r] * inv);
        ot[(mb + r) * 64 + wc + 16 + n] = (_Float16)(o1[r] * inv);
      }
    }
    __syncthreads();
    // Wo projection: Yh[ipg][32] = wos[32][64] @ ot^T, one tile per wave
    {
      const int mt = w >> 1, nt = w & 1;
      v8f cy = v8zero();
      cy = wmma_f16(frag_a_f16(wos, mt * 16, 64),      frag_b_f16_T(ot, nt * 16, 64), cy);
      cy = wmma_f16(frag_a_f16(wos + 32, mt * 16, 64), frag_b_f16_T(ot + 32, nt * 16, 64), cy);
      const int n = nt * 16 + (l & 15);
      const int mb2 = mt * 16 + ((l >> 4) << 3);
#pragma unroll
      for (int r = 0; r < 8; ++r) {
        const int m = mb2 + r;
        if (m < ipg)
          Y[((size_t)seq * C + head * ipg + m) * L + r0 + n] = cy[r] + Bo[head * ipg + m];
      }
    }
    __syncthreads();
  }
}

// ---------------------------------------------------------------------------
// BatchNorm (training-mode batch stats over (batch, L)) : stats + apply
// ---------------------------------------------------------------------------
#define BN_RES  1
#define BN_RELU 2

__global__ void k_fill(float* p, int n, float v) {
  const int i = blockIdx.x * blockDim.x + threadIdx.x;
  if (i < n) p[i] = v;
}

__global__ __launch_bounds__(256) void k_bn_stats(
    const float* __restrict__ X, float* __restrict__ stats, int NB, int C, int L)
{
  const int cch = blockIdx.x;
  const long long total = (long long)NB * L;
  float s = 0.f, s2 = 0.f;
  for (long long i = (long long)blockIdx.y * blockDim.x + threadIdx.x; i < total;
       i += (long long)gridDim.y * blockDim.x) {
    const long long nb2 = i / L, ll = i % L;
    const float v = X[(nb2 * C + cch) * L + ll];
    s += v; s2 += v * v;
  }
  __shared__ float r1[256], r2[256];
  const int t = threadIdx.x;
  r1[t] = s; r2[t] = s2;
  __syncthreads();
  for (int sh = 128; sh > 0; sh >>= 1) {
    if (t < sh) { r1[t] += r1[t + sh]; r2[t] += r2[t + sh]; }
    __syncthreads();
  }
  if (t == 0) { atomicAdd(&stats[cch], r1[0]); atomicAdd(&stats[C + cch], r2[0]); }
}

__global__ void k_bn_apply(float* __restrict__ Y, const float* __restrict__ X,
                           const float* __restrict__ R, const float* __restrict__ stats,
                           long long total, int C, int L, float invcnt, int flags)
{
  for (long long i = (long long)blockIdx.x * blockDim.x + threadIdx.x; i < total;
       i += (long long)gridDim.x * blockDim.x) {
    const int cch = (int)((i / L) % C);
    const float m = stats[cch] * invcnt;
    const float var = stats[C + cch] * invcnt - m * m;
    float y = (X[i] - m) * rsqrtf(var + 1e-5f);
    if (flags & BN_RES) y += R[i];
    if (flags & BN_RELU) y = fmaxf(y, 0.f);
    Y[i] = y;
  }
}

__global__ void k_add_relu(float* __restrict__ o, const float* __restrict__ a,
                           const float* __restrict__ b, long long n) {
  for (long long i = (long long)blockIdx.x * blockDim.x + threadIdx.x; i < n;
       i += (long long)gridDim.x * blockDim.x)
    o[i] = fmaxf(a[i] + b[i], 0.f);
}

__global__ void k_copy(float* __restrict__ o, const float* __restrict__ a, long long n) {
  for (long long i = (long long)blockIdx.x * blockDim.x + threadIdx.x; i < n;
       i += (long long)gridDim.x * blockDim.x)
    o[i] = a[i];
}

// ---------------------------------------------------------------------------
// Point-cloud support kernels
// ---------------------------------------------------------------------------
__global__ void k_xyzT(float* __restrict__ xyz, const float* __restrict__ x, int B_, int P) {
  const int i = blockIdx.x * blockDim.x + threadIdx.x;
  if (i >= B_ * P) return;
  const int p = i % P, b = i / P;
  for (int d = 0; d < 3; ++d)
    xyz[((size_t)b * P + p) * 3 + d] = x[((size_t)b * 3 + d) * P + p];
}

__global__ __launch_bounds__(1024) void k_fps(const float* __restrict__ xyz,
                                              int* __restrict__ fidx, int P, int G)
{
  __shared__ float bd[1024];
  __shared__ int bi[1024];
  __shared__ float lp[3];
  const int b = blockIdx.x, t = threadIdx.x;
  const float* xb = xyz + (size_t)b * P * 3;
  float px = 0.f, py = 0.f, pz = 0.f, dist = 3.0e38f;
  if (t < P) { px = xb[t * 3]; py = xb[t * 3 + 1]; pz = xb[t * 3 + 2]; }
  int last = 0;
  for (int i = 0; i < G; ++i) {
    if (t == 0) {
      fidx[(size_t)b * G + i] = last;
      lp[0] = xb[last * 3]; lp[1] = xb[last * 3 + 1]; lp[2] = xb[last * 3 + 2];
    }
    __syncthreads();
    if (t < P) {
      const float dx = px - lp[0], dy = py - lp[1], dz = pz - lp[2];
      dist = fminf(dist, dx * dx + dy * dy + dz * dz);
      bd[t] = dist;
    } else bd[t] = -1.f;
    bi[t] = t;
    __syncthreads();
    for (int sh = 512; sh > 0; sh >>= 1) {
      if (t < sh) {
        if (bd[t + sh] > bd[t] || (bd[t + sh] == bd[t] && bi[t + sh] < bi[t])) {
          bd[t] = bd[t + sh]; bi[t] = bi[t + sh];
        }
      }
      __syncthreads();
    }
    last = bi[0];
    __syncthreads();
  }
}

__global__ __launch_bounds__(256) void k_knn(const float* __restrict__ xyz,
    const int* __restrict__ fidx, int* __restrict__ kidx, int P, int G)
{
  __shared__ float d[1024];
  __shared__ float rb[256];
  __shared__ int ri[256];
  const int bgi = blockIdx.x, b = bgi / G, t = threadIdx.x;
  const float* xb = xyz + (size_t)b * P * 3;
  const int ci = fidx[bgi];
  const float cx = xb[ci * 3], cy = xb[ci * 3 + 1], cz = xb[ci * 3 + 2];
  for (int p = t; p < P; p += 256) {
    const float dx = xb[p * 3] - cx, dy = xb[p * 3 + 1] - cy, dz = xb[p * 3 + 2] - cz;
    d[p] = dx * dx + dy * dy + dz * dz;
  }
  __syncthreads();
  for (int kk = 0; kk < 32; ++kk) {
    float best = 3.0e38f; int bidx = P;
    for (int p = t; p < P; p += 256)
      if (d[p] < best) { best = d[p]; bidx = p; }
    rb[t] = best; ri[t] = bidx;
    __syncthreads();
    for (int sh = 128; sh > 0; sh >>= 1) {
      if (t < sh) {
        if (rb[t + sh] < rb[t] || (rb[t + sh] == rb[t] && ri[t + sh] < ri[t])) {
          rb[t] = rb[t + sh]; ri[t] = ri[t + sh];
        }
      }
      __syncthreads();
    }
    if (t == 0) { kidx[(size_t)bgi * 32 + kk] = ri[0]; d[ri[0]] = 3.0e38f; }
    __syncthreads();
  }
}

// new_xyz + normalized aug[bg][32][10] + gpts[bg][2C][32]
__global__ __launch_bounds__(32) void k_group(const float* __restrict__ xyz,
    const float* __restrict__ F, const int* __restrict__ fidx, const int* __restrict__ kidx,
    float* __restrict__ nxyz, float* __restrict__ aug, float* __restrict__ gp,
    int P, int G, int Cl)
{
  const int bgi = blockIdx.x, b = bgi / G, t = threadIdx.x;
  const int ci = fidx[bgi];
  const float* xb = xyz + (size_t)b * P * 3;
  const float ax = xb[ci * 3], ay = xb[ci * 3 + 1], az = xb[ci * 3 + 2];
  if (t == 0) {
    nxyz[(size_t)bgi * 3] = ax; nxyz[(size_t)bgi * 3 + 1] = ay; nxyz[(size_t)bgi * 3 + 2] = az;
  }
  const int p = kidx[(size_t)bgi * 32 + t];
  const float gx = xb[p * 3], gy = xb[p * 3 + 1], gz = xb[p * 3 + 2];
  const float rx = gx - ax, ry = gy - ay, rz = gz - az;
  float f[10] = { sqrtf(rx * rx + ry * ry + rz * rz), rx, ry, rz, ax, ay, az, gx, gy, gz };
  __shared__ float sf[10][32];
  __shared__ float mean[10], inv[10];
  for (int i = 0; i < 10; ++i) sf[i][t] = f[i];
  __syncthreads();
  if (t < 10) {
    float m = 0.f; for (int j = 0; j < 32; ++j) m += sf[t][j]; m *= (1.f / 32.f);
    float v = 0.f; for (int j = 0; j < 32; ++j) { const float d = sf[t][j] - m; v += d * d; }
    mean[t] = m; inv[t] = 1.f / (sqrtf(v / 31.f) + 1e-8f);
  }
  __syncthreads();
  float* ao = aug + ((size_t)bgi * 32 + t) * 10;
  for (int i = 0; i < 10; ++i) ao[i] = (f[i] - mean[i]) * inv[i];
  const float* Fb = F + (size_t)b * Cl * P;
  for (int cc = 0; cc < Cl; ++cc) {
    gp[((size_t)bgi * 2 * Cl + cc) * 32 + t]      = Fb[(size_t)cc * P + p];
    gp[((size_t)bgi * 2 * Cl + Cl + cc) * 32 + t] = Fb[(size_t)cc * P + ci];
  }
}

// pos_extract augmentation (with the reference's std/mean swap), aug[b][G][10]
__global__ __launch_bounds__(256) void k_posaug(const float* __restrict__ xyz,
    const float* __restrict__ alpha, const float* __restrict__ beta,
    float* __restrict__ aug, int G)
{
  const int b = blockIdx.x, t = threadIdx.x;
  __shared__ float red[256];
  __shared__ float mean3[3], std3[3];
  const float* xb = xyz + (size_t)b * G * 3;
  for (int ch = 0; ch < 3; ++ch) {
    float s = 0.f;
    for (int p = t; p < G; p += 256) s += xb[p * 3 + ch];
    red[t] = s; __syncthreads();
    for (int sh = 128; sh > 0; sh >>= 1) { if (t < sh) red[t] += red[t + sh]; __syncthreads(); }
    const float m = red[0] / (float)G;
    __syncthreads();
    float v = 0.f;
    for (int p = t; p < G; p += 256) { const float d = xb[p * 3 + ch] - m; v += d * d; }
    red[t] = v; __syncthreads();
    for (int sh = 128; sh > 0; sh >>= 1) { if (t < sh) red[t] += red[t + sh]; __syncthreads(); }
    if (t == 0) { mean3[ch] = m; std3[ch] = sqrtf(red[0] / (float)(G - 1)); }
    __syncthreads();
  }
  for (int p = t; p < G; p += 256) {
    const float px = xb[p * 3], py = xb[p * 3 + 1], pz = xb[p * 3 + 2];
    float f[10];
    f[0] = px; f[1] = py; f[2] = pz;
    f[3] = (px - std3[0]) / (mean3[0] + 1e-8f);   // faithful std/mean swap
    f[4] = (py - std3[1]) / (mean3[1] + 1e-8f);
    f[5] = (pz - std3[2]) / (mean3[2] + 1e-8f);
    f[6] = sqrtf(px * px + py * py + pz * pz);
    f[7] = cosf(alpha[0] * px + beta[0]);
    f[8] = cosf(alpha[1] * py + beta[1]);
    f[9] = cosf(alpha[2] * pz + beta[2]);
    float m = 0.f; for (int i = 0; i < 10; ++i) m += f[i]; m *= 0.1f;
    float v = 0.f; for (int i = 0; i < 10; ++i) { const float d = f[i] - m; v += d * d; }
    const float inv = 1.f / (sqrtf(v / 9.f) + 1e-8f);
    float* o = aug + ((size_t)b * G + p) * 10;
    for (int i = 0; i < 10; ++i) o[i] = (f[i] - m) * inv;
  }
}

// pos_embed MLP: 10 -> 40 -> C, one block / point; pe[seq][C][L]
__global__ __launch_bounds__(64) void k_posmlp(const float* __restrict__ aug,
    const float* __restrict__ w1, const float* __restrict__ b1,
    const float* __restrict__ w2, const float* __restrict__ b2,
    float* __restrict__ pe, int C, int L)
{
  __shared__ float a[10];
  __shared__ float h[40];
  const int pt = blockIdx.x, t = threadIdx.x;
  if (t < 10) a[t] = aug[(size_t)pt * 10 + t];
  __syncthreads();
  if (t < 40) {
    float s = b1[t];
    for (int i = 0; i < 10; ++i) s += a[i] * w1[i * 40 + t];
    h[t] = fmaxf(s, 0.f);
  }
  __syncthreads();
  const int seq = pt / L, ll = pt % L;
  for (int o = t; o < C; o += 64) {
    float s = b2[o];
    for (int j = 0; j < 40; ++j) s += h[j] * w2[(size_t)j * C + o];
    pe[((size_t)seq * C + o) * L + ll] = s;
  }
}

// max over L: out[b][C][G] = max_l x[(b*G+g)][C][l]
__global__ void k_maxpool(float* __restrict__ out, const float* __restrict__ xin,
                          int B_, int C, int G, int L)
{
  const int i = blockIdx.x * blockDim.x + threadIdx.x;
  if (i >= B_ * C * G) return;
  const int g = i % G, cc = (i / G) % C, b = i / (G * C);
  const float* p = xin + ((size_t)(b * G + g) * C + cc) * L;
  float m = p[0];
  for (int ll = 1; ll < L; ++ll) m = fmaxf(m, p[ll]);
  out[((size_t)b * C + cc) * G + g] = m;
}

// ---------------------------------------------------------------------------
// Host orchestration
// ---------------------------------------------------------------------------
extern "C" void kernel_launch(void* const* d_in, const int* in_sizes, int n_in,
                              void* d_out, int out_size, void* d_ws, size_t ws_size,
                              hipStream_t stream)
{
  (void)in_sizes; (void)out_size; (void)ws_size;
  const float* x_in = (const float*)d_in[0];
  int pi = 1;
  auto nx = [&]() -> const float* { return (pi < n_in) ? (const float*)d_in[pi++] : nullptr; };

  struct BlockP { const float *bo,*f1b,*f1w,*f2b,*f2w,*wk,*wo,*wq,*wv; };
  struct PEp { const float *b1,*b2,*w1,*w2; };
  struct Sub { const float *alpha,*beta; BlockP blk[2]; PEp pe; const float *fc_b,*fc_w; };
  Sub pre[4], pos[4];
  // jax pytree flatten order: dicts sorted by key.
  const float* emb1_b = nx(); const float* emb1_w = nx();
  const float* emb2_b = nx(); const float* emb2_w = nx();
  auto rdblk = [&](BlockP& bp) {
    bp.bo = nx(); bp.f1b = nx(); bp.f1w = nx(); bp.f2b = nx(); bp.f2w = nx();
    bp.wk = nx(); bp.wo = nx(); bp.wq = nx(); bp.wv = nx();
  };
  for (int s = 0; s < 4; ++s) {
    // 'pos' < 'pre'
    pos[s].alpha = nx(); pos[s].beta = nx();
    rdblk(pos[s].blk[0]); rdblk(pos[s].blk[1]);
    pos[s].pe.b1 = nx(); pos[s].pe.b2 = nx(); pos[s].pe.w1 = nx(); pos[s].pe.w2 = nx();
    rdblk(pre[s].blk[0]); rdblk(pre[s].blk[1]);
    pre[s].fc_b = nx(); pre[s].fc_w = nx();
    pre[s].pe.b1 = nx(); pre[s].pe.b2 = nx(); pre[s].pe.w1 = nx(); pre[s].pe.w2 = nx();
  }

  // workspace bump allocator
  char* base = (char*)d_ws;
  size_t off = 0;
  auto alloc = [&](size_t nbytes) -> void* {
    off = (off + 255) & ~(size_t)255;
    void* p = base + off; off += nbytes; return p;
  };
  const size_t BIG = (size_t)2 * 1024 * 1024;     // elements (8MB f32)
  float* xr    = (float*)alloc(BIG * 4);
  float* att   = (float*)alloc(BIG * 4);
  float* tmp1  = (float*)alloc(BIG * 4);
  float* tmp2  = (float*)alloc(BIG * 4);
  float* pebuf = (float*)alloc(BIG * 4);
  float* gpts  = (float*)alloc(BIG * 4);
  float* augb  = (float*)alloc((size_t)1024 * 2 * 32 * 10 * 4);
  float* posag = (float*)alloc((size_t)2 * 512 * 10 * 4);
  float* fA    = (float*)alloc((size_t)2 * 32 * 1024 * 4);
  float* fB    = (float*)alloc((size_t)2 * 32 * 1024 * 4);
  float* xyzA  = (float*)alloc((size_t)2 * 1024 * 3 * 4);
  float* xyzB  = (float*)alloc((size_t)2 * 1024 * 3 * 4);
  float* stats = (float*)alloc((size_t)1024 * 4);
  int*   fidx  = (int*)alloc((size_t)2 * 512 * 4);
  int*   kidx  = (int*)alloc((size_t)2 * 512 * 32 * 4);

  auto gemm = [&](float* Y, const float* X, const float* W, const float* bias,
                  int nb, int M, int K, int N, int flags) {
    dim3 g(nb, (M + 63) / 64, N / 32);
    k_wmma_gemm<<<g, 128, 0, stream>>>(Y, X, W, bias, M, K, N, flags);
  };
  auto bn = [&](const float* xin, float* xout, const float* res,
                int nb, int C, int L, int flags) {
    k_fill<<<(2 * C + 255) / 256, 256, 0, stream>>>(stats, 2 * C, 0.f);
    k_bn_stats<<<dim3(C, 8), 256, 0, stream>>>(xin, stats, nb, C, L);
    const long long total = (long long)nb * C * L;
    k_bn_apply<<<(int)((total + 255) / 256), 256, 0, stream>>>(
        xout, xin, res, stats, total, C, L, 1.f / (float)((long long)nb * L), flags);
  };
  auto attn_smem = [&](int C, int L) -> size_t {
    (void)C;
    const size_t nf32 = (size_t)32 * L + 1024 + 96;
    const size_t nf16 = (size_t)3 * L * 64 + 3 * 64 * 32 + 32 * 64 + 1024 + 32 * 64;
    return nf32 * 4 + nf16 * 2;
  };
  auto tblock = [&](float* xb, const float* peb, const BlockP& bp, int nseq, int C, int L) {
    const size_t sm = attn_smem(C, L);
    k_attn_fused<<<dim3(nseq, 16), 128, sm, stream>>>(
        tmp1, xb, peb, bp.wq, bp.wk, bp.wv, bp.wo, bp.bo, C, L);
    bn(tmp1, att, xb, nseq, C, L, BN_RES | BN_RELU);          // att = relu(bn(attn)+x)
    const int h = C >> 3;
    gemm(tmp1, att, bp.f1w, bp.f1b, nseq, h, C, L, GF_BIAS);
    bn(tmp1, tmp2, nullptr, nseq, h, L, BN_RELU);             // hidden
    gemm(tmp1, tmp2, bp.f2w, bp.f2b, nseq, C, h, L, GF_BIAS);
    bn(tmp1, tmp2, nullptr, nseq, C, L, BN_RELU);             // ffn out
    const long long total = (long long)nseq * C * L;
    k_add_relu<<<(int)((total + 255) / 256), 256, 0, stream>>>(xb, att, tmp2, total);
  };

  // ---- embedding: 3 -> 32 -> 32 ----
  gemm(tmp1, x_in, emb1_w, emb1_b, 2, 32, 3, 1024, GF_BIAS);
  bn(tmp1, fA, nullptr, 2, 32, 1024, BN_RELU);
  gemm(tmp1, fA, emb2_w, emb2_b, 2, 32, 32, 1024, GF_BIAS);
  bn(tmp1, fB, nullptr, 2, 32, 1024, BN_RELU);
  float* fcur = fB; float* fnext = fA;
  k_xyzT<<<(2 * 1024 + 255) / 256, 256, 0, stream>>>(xyzA, x_in, 2, 1024);
  float* xyzcur = xyzA; float* xyznext = xyzB;

  const int CHAN[4] = {64, 128, 256, 512};
  const int GRP[4]  = {512, 256, 128, 64};
  int P = 1024, lastC = 32;
  for (int s = 0; s < 4; ++s) {
    const int C = CHAN[s], G = GRP[s], bg = 2 * G;
    k_fps<<<2, 1024, 0, stream>>>(xyzcur, fidx, P, G);
    k_knn<<<bg, 256, 0, stream>>>(xyzcur, fidx, kidx, P, G);
    k_group<<<bg, 32, 0, stream>>>(xyzcur, fcur, fidx, kidx, xyznext, augb, gpts, P, G, lastC);
    // pre_extract over bg sequences of L=32
    gemm(tmp1, gpts, pre[s].fc_w, pre[s].fc_b, bg, C, 2 * lastC, 32, GF_BIAS);
    bn(tmp1, xr, nullptr, bg, C, 32, BN_RELU);
    k_posmlp<<<bg * 32, 64, 0, stream>>>(augb, pre[s].pe.w1, pre[s].pe.b1,
                                         pre[s].pe.w2, pre[s].pe.b2, pebuf, C, 32);
    tblock(xr, pebuf, pre[s].blk[0], bg, C, 32);
    tblock(xr, pebuf, pre[s].blk[1], bg, C, 32);
    k_maxpool<<<(2 * C * G + 255) / 256, 256, 0, stream>>>(fnext, xr, 2, C, G, 32);
    // pos_extract over 2 sequences of L=G
    k_posaug<<<2, 256, 0, stream>>>(xyznext, pos[s].alpha, pos[s].beta, posag, G);
    k_posmlp<<<2 * G, 64, 0, stream>>>(posag, pos[s].pe.w1, pos[s].pe.b1,
                                       pos[s].pe.w2, pos[s].pe.b2, pebuf, C, G);
    tblock(fnext, pebuf, pos[s].blk[0], 2, C, G);
    tblock(fnext, pebuf, pos[s].blk[1], 2, C, G);
    // rotate buffers
    float* tf = fcur; fcur = fnext; fnext = tf;
    float* tx = xyzcur; xyzcur = xyznext; xyznext = tx;
    P = G; lastC = C;
  }
  const long long total = (long long)2 * 512 * 64;
  k_copy<<<(int)((total + 255) / 256), 256, 0, stream>>>((float*)d_out, fcur, total);
}
